// MoEModel_87849261073059
// MI455X (gfx1250) — matrix operand order repaired and checked by reading, not compiled, not run on hardware
//
#include <hip/hip_runtime.h>
#include <hip/hip_bf16.h>

#define BB 32
#define TT 512
#define GG 256
#define EE 8
#define VV 8192
#define DD 512
#define HH 2048

#define TM 16          // token rows per block
#define NCHUNK 256     // N columns per chunk
#define KSTEP 64       // K rows staged per step
#define WSTR 72        // sW row stride in bf16 (padded, multiple of 8)

typedef __attribute__((ext_vector_type(16))) __bf16 v16bf;
typedef __attribute__((ext_vector_type(8)))  __bf16 v8bf;
typedef __attribute__((ext_vector_type(8)))  float  v8f;

// ---------- helpers ----------

__device__ __forceinline__ float fast_tanh(float u) {
#if __has_builtin(__builtin_amdgcn_tanhf)
    return __builtin_amdgcn_tanhf(u);
#elif __has_builtin(__builtin_amdgcn_tanh_f32)
    return __builtin_amdgcn_tanh_f32(u);
#else
    // branch-free: tanh(u) = (e^{2u}-1)/(e^{2u}+1), clamped so v_exp never overflows
    u = fminf(fmaxf(u, -9.0f), 9.0f);
    float e = __expf(2.0f * u);
    return (e - 1.0f) * __frcp_rn(e + 1.0f);
#endif
}

__device__ __forceinline__ float gelu_tanh(float x) {
    float u = 0.7978845608028654f * (x + 0.044715f * x * x * x);
    return 0.5f * x * (1.0f + fast_tanh(u));
}

// A-operand (16x32 bf16 tile) from LDS row-major [16][stride]
// lane<16: row=lane, K = k0+0..7 then k0+16..23 ; lane>=16: row=lane-16, K = k0+8..15 then k0+24..31
__device__ __forceinline__ v16bf load_a(const __bf16* base, int stride, int k0, int lane) {
    int r  = lane & 15;
    int hi = lane >> 4;
    const __bf16* p = base + r * stride + k0 + hi * 8;
    v8bf c0 = *(const v8bf*)(p);
    v8bf c1 = *(const v8bf*)(p + 16);
    v16bf a;
#pragma unroll
    for (int i = 0; i < 8; ++i) { a[i] = c0[i]; a[8 + i] = c1[i]; }
    return a;
}

// B-operand (32x16 bf16 tile) from LDS staged transposed [N][WSTR]
// lane<16: col=nLocal+lane, K=kLocal+0..7 then kLocal+16..23 ; lane>=16: col=nLocal+lane-16, K=kLocal+8..15 then kLocal+24..31
__device__ __forceinline__ v16bf load_b(const __bf16* sW, int nLocal, int kLocal, int lane) {
    int n  = nLocal + (lane & 15);
    int hi = lane >> 4;
    const __bf16* p = sW + n * WSTR + kLocal + hi * 8;
    v8bf c0 = *(const v8bf*)(p);
    v8bf c1 = *(const v8bf*)(p + 16);
    v16bf b;
#pragma unroll
    for (int i = 0; i < 8; ++i) { b[i] = c0[i]; b[8 + i] = c1[i]; }
    return b;
}

// Stage KSTEP x NCHUNK fp32 tile of W (row-major, leading dim ld) into sW as bf16,
// transposed to [N][WSTR]. Each thread reads two consecutive K rows (coalesced
// float4 along N) and writes (k,k+1) bf16 pairs as packed u32 -> ds_store_b32.
// Also issues a global_prefetch of the next K tile.
__device__ __forceinline__ void stage_w(const float* __restrict__ W, int ld,
                                        int kk, int n0, int kmax,
                                        __bf16* sW, int tid) {
#pragma unroll
    for (int i = 0; i < 8; ++i) {
        int idx = i * 256 + tid;         // 0..2047
        int kp  = idx >> 6;              // K-pair index 0..31
        int n4  = idx & 63;              // float4 index along N
        const float* base = W + (size_t)(kk + 2 * kp) * ld + n0 + n4 * 4;
        float4 r0 = *(const float4*)(base);
        float4 r1 = *(const float4*)(base + ld);
        __bf16* d = sW + (n4 * 4) * WSTR + 2 * kp;
        union { __bf16 h[2]; unsigned int u; } p;
        p.h[0] = (__bf16)r0.x; p.h[1] = (__bf16)r1.x; *(unsigned int*)(d)            = p.u;
        p.h[0] = (__bf16)r0.y; p.h[1] = (__bf16)r1.y; *(unsigned int*)(d + WSTR)     = p.u;
        p.h[0] = (__bf16)r0.z; p.h[1] = (__bf16)r1.z; *(unsigned int*)(d + 2 * WSTR) = p.u;
        p.h[0] = (__bf16)r0.w; p.h[1] = (__bf16)r1.w; *(unsigned int*)(d + 3 * WSTR) = p.u;
    }
    if (kk + KSTEP < kmax) {
        // one prefetch line per thread into the next K tile (global_prefetch_b8)
        const float* np = W + (size_t)(kk + KSTEP + (tid >> 2)) * ld + n0 + (tid & 3) * 64;
        __builtin_prefetch(np, 0, 3);
    }
}

// ---------- kernel 1: gating ----------
__global__ void gate_kernel(const float* __restrict__ gf, const float* __restrict__ Wg,
                            const float* __restrict__ bg, float* __restrict__ out,
                            int* __restrict__ ws_assign, float* __restrict__ ws_nll) {
    int b = threadIdx.x;
    if (b >= BB) return;
    float acc[EE];
#pragma unroll
    for (int e = 0; e < EE; ++e) acc[e] = bg[e];
    for (int g = 0; g < GG; ++g) {
        float x = gf[b * GG + g];
        const float* wr = Wg + g * EE;
#pragma unroll
        for (int e = 0; e < EE; ++e) acc[e] += x * wr[e];
    }
    float m = acc[0]; int am = 0;
#pragma unroll
    for (int e = 1; e < EE; ++e) if (acc[e] > m) { m = acc[e]; am = e; }
    float s = 0.f, ex[EE];
#pragma unroll
    for (int e = 0; e < EE; ++e) { ex[e] = __expf(acc[e] - m); s += ex[e]; }
#pragma unroll
    for (int e = 0; e < EE; ++e) out[1 + BB + b * EE + e] = ex[e] / s;
    out[1 + b] = (float)am;
    ws_assign[b] = am;
    ws_nll[b] = 0.f;
}

// ---------- kernel 2: fused expert FFN + streaming log-softmax NLL ----------
__global__ __launch_bounds__(256)
void moe_fused(const int* __restrict__ ids, const int* __restrict__ tgt,
               const float* __restrict__ Emb, const float* __restrict__ W1,
               const float* __restrict__ b1, const float* __restrict__ W2,
               const float* __restrict__ b2, const int* __restrict__ ws_assign,
               float* __restrict__ ws_nll) {
    extern __shared__ char smem[];
    __bf16* sA   = (__bf16*)(smem);                     // 16*512*2  = 16384
    __bf16* sH   = (__bf16*)(smem + 16384);             // 16*2048*2 = 65536
    __bf16* sW   = (__bf16*)(smem + 16384 + 65536);     // 256*72*2  = 36864
    float*  sLog = (float*) (smem + 118784);            // 16*256*4  = 16384
    float*  sMax = (float*) (smem + 135168);            // 16*4
    float*  sSum = sMax + TM;
    float*  sTgt = sSum + TM;
    int*    sTc  = (int*)(sTgt + TM);

    const int tid  = threadIdx.x;
    const int lane = tid & 31;
    const int w    = tid >> 5;             // wave id 0..7
    const int b    = blockIdx.y;
    const int t0   = blockIdx.x * TM;
    const int e    = ws_assign[b];

    const float* Embe = Emb + (size_t)e * VV * DD;
    const float* W1e  = W1  + (size_t)e * DD * HH;
    const float* b1e  = b1  + (size_t)e * HH;
    const float* W2e  = W2  + (size_t)e * HH * VV;
    const float* b2e  = b2  + (size_t)e * VV;

    // ---- gather embedding rows -> sA (bf16) ----
#pragma unroll
    for (int i = 0; i < 8; ++i) {
        int idx = i * 256 + tid;           // 0..2047 float4 slots (16 rows x 128)
        int r   = idx >> 7;
        int c4  = idx & 127;
        int id  = ids[b * TT + t0 + r];
        const float4 v = *(const float4*)(Embe + (size_t)id * DD + c4 * 4);
        __bf16* d = sA + r * DD + c4 * 4;
        d[0] = (__bf16)v.x; d[1] = (__bf16)v.y; d[2] = (__bf16)v.z; d[3] = (__bf16)v.w;
    }
    __syncthreads();

    const int r0 = lane >> 4;
    const int nl = lane & 15;

    // ---- GEMM1: h = gelu(sA(16x512) @ W1e(512x2048) + b1) -> sH bf16 ----
    for (int n1 = 0; n1 < HH; n1 += NCHUNK) {
        v8f acc0 = {}; v8f acc1 = {};
        for (int kk = 0; kk < DD; kk += KSTEP) {
            __syncthreads();
            stage_w(W1e, HH, kk, n1, DD, sW, tid);
            __syncthreads();
#pragma unroll
            for (int ks = 0; ks < KSTEP; ks += 32) {
                v16bf a  = load_a(sA, DD, kk + ks, lane);
                v16bf bm0 = load_b(sW, (2 * w) * 16,     ks, lane);
                acc0 = __builtin_amdgcn_wmma_f32_16x16x32_bf16(false, a, false, bm0,
                                                               (short)0, acc0, false, false);
                v16bf bm1 = load_b(sW, (2 * w + 1) * 16, ks, lane);
                acc1 = __builtin_amdgcn_wmma_f32_16x16x32_bf16(false, a, false, bm1,
                                                               (short)0, acc1, false, false);
            }
        }
#pragma unroll
        for (int j = 0; j < 2; ++j) {
            int col = n1 + (2 * w + j) * 16 + nl;
            float bias = b1e[col];
            v8f acc = j ? acc1 : acc0;
#pragma unroll
            for (int i = 0; i < 8; ++i) {
                int row = i + r0 * 8;
                sH[row * HH + col] = (__bf16)gelu_tanh(acc[i] + bias);
            }
        }
    }

    // ---- init online-softmax state ----
    __syncthreads();
    if (tid < TM) {
        sMax[tid] = -3.4e38f;
        sSum[tid] = 0.f;
        sTgt[tid] = 0.f;
        sTc[tid]  = tgt[b * TT + t0 + tid];
    }
    __syncthreads();

    // ---- GEMM2 + streaming log-softmax over V ----
    for (int n2 = 0; n2 < VV; n2 += NCHUNK) {
        v8f acc0 = {}; v8f acc1 = {};
        for (int kk = 0; kk < HH; kk += KSTEP) {
            __syncthreads();
            stage_w(W2e, VV, kk, n2, HH, sW, tid);
            __syncthreads();
#pragma unroll
            for (int ks = 0; ks < KSTEP; ks += 32) {
                v16bf a  = load_a(sH, HH, kk + ks, lane);
                v16bf bm0 = load_b(sW, (2 * w) * 16,     ks, lane);
                acc0 = __builtin_amdgcn_wmma_f32_16x16x32_bf16(false, a, false, bm0,
                                                               (short)0, acc0, false, false);
                v16bf bm1 = load_b(sW, (2 * w + 1) * 16, ks, lane);
                acc1 = __builtin_amdgcn_wmma_f32_16x16x32_bf16(false, a, false, bm1,
                                                               (short)0, acc1, false, false);
            }
        }
        // logits chunk -> sLog (with b2 bias)
#pragma unroll
        for (int j = 0; j < 2; ++j) {
            int col = n2 + (2 * w + j) * 16 + nl;
            float bias = b2e[col];
            v8f acc = j ? acc1 : acc0;
#pragma unroll
            for (int i = 0; i < 8; ++i) {
                int row = i + r0 * 8;
                sLog[row * NCHUNK + (col - n2)] = acc[i] + bias;
            }
        }
        __syncthreads();
        // online softmax update: wave w owns rows 2w, 2w+1
#pragma unroll
        for (int rr = 0; rr < 2; ++rr) {
            int r = 2 * w + rr;
            float lm = -3.4e38f;
#pragma unroll
            for (int i = 0; i < 8; ++i) lm = fmaxf(lm, sLog[r * NCHUNK + lane + i * 32]);
#pragma unroll
            for (int off = 16; off >= 1; off >>= 1) lm = fmaxf(lm, __shfl_xor(lm, off, 32));
            float prev = sMax[r];
            float nm = fmaxf(prev, lm);
            float ls = 0.f;
#pragma unroll
            for (int i = 0; i < 8; ++i) ls += __expf(sLog[r * NCHUNK + lane + i * 32] - nm);
#pragma unroll
            for (int off = 16; off >= 1; off >>= 1) ls += __shfl_xor(ls, off, 32);
            if (lane == 0) {
                sSum[r] = sSum[r] * __expf(prev - nm) + ls;
                sMax[r] = nm;
                int tc = sTc[r] - n2;
                if (tc >= 0 && tc < NCHUNK) sTgt[r] = sLog[r * NCHUNK + tc];
            }
        }
        __syncthreads();
    }

    // ---- per-block NLL partial ----
    if (w == 0) {
        float nll = 0.f;
        if (lane < TM) nll = -(sTgt[lane] - (sMax[lane] + __logf(sSum[lane])));
#pragma unroll
        for (int off = 16; off >= 1; off >>= 1) nll += __shfl_xor(nll, off, 32);
        if (lane == 0) atomicAdd(&ws_nll[b], nll);
    }
}

// ---------- kernel 3: segment-mean total loss ----------
__global__ void finalize_kernel(const int* __restrict__ ws_assign,
                                const float* __restrict__ ws_nll,
                                float* __restrict__ out) {
    if (threadIdx.x == 0) {
        float sums[EE] = {}; float cnt[EE] = {};
        for (int b = 0; b < BB; ++b) {
            float ps = ws_nll[b] / (float)TT;
            int e = ws_assign[b];
            sums[e] += ps; cnt[e] += 1.f;
        }
        float tot = 0.f;
#pragma unroll
        for (int e = 0; e < EE; ++e) if (cnt[e] > 0.f) tot += sums[e] / cnt[e];
        out[0] = tot;
    }
}

extern "C" void kernel_launch(void* const* d_in, const int* in_sizes, int n_in,
                              void* d_out, int out_size, void* d_ws, size_t ws_size,
                              hipStream_t stream) {
    const float* gf  = (const float*)d_in[0];
    const int*   ids = (const int*)  d_in[1];
    const int*   tgt = (const int*)  d_in[2];
    const float* Wg  = (const float*)d_in[3];
    const float* bg  = (const float*)d_in[4];
    const float* Emb = (const float*)d_in[5];
    const float* W1  = (const float*)d_in[6];
    const float* b1  = (const float*)d_in[7];
    const float* W2  = (const float*)d_in[8];
    const float* b2  = (const float*)d_in[9];
    float* out = (float*)d_out;

    int*   ws_assign = (int*)d_ws;
    float* ws_nll    = (float*)((char*)d_ws + 128);

    gate_kernel<<<1, 32, 0, stream>>>(gf, Wg, bg, out, ws_assign, ws_nll);

    dim3 grid(TT / TM, BB);                 // 32 x 32 blocks
    size_t shmem = 135168 + (3 * TM * 4) + (TM * 4);  // 135424 bytes
    moe_fused<<<grid, 256, shmem, stream>>>(ids, tgt, Emb, W1, b1, W2, b2,
                                            ws_assign, ws_nll);

    finalize_kernel<<<1, 32, 0, stream>>>(ws_assign, ws_nll, out);
}
